// SpecialTokenCompressedLMHead_66125316489776
// MI455X (gfx1250) — matrix operand order repaired and checked by reading, not compiled, not run on hardware
//
#include <hip/hip_runtime.h>
#include <stdint.h>

// ---------------- problem constants (from the reference) ----------------
#define BB   2
#define SS   1024
#define DD   1024              // token dim (GEMM K)
#define MM   (BB * SS)         // 2048 rows
#define VCC  8192              // compressed vocab (GEMM N)
#define VV   32000             // full vocab
#define KH   4                 // hash functions
#define KT   32                // GEMM K step (one bf16 WMMA depth)
#define LDA  32                // LDS row stride in ushorts (64 B)

typedef __attribute__((ext_vector_type(16))) __bf16 v16bf;
typedef __attribute__((ext_vector_type(8)))  float  v8f;

// ---------------- CDNA5 async global->LDS copy (ASYNCcnt) ----------------
#if defined(__gfx1250__) && __has_builtin(__builtin_amdgcn_global_load_async_to_lds_b128) && \
    __has_builtin(__builtin_amdgcn_s_wait_asynccnt)
typedef int v4i_t __attribute__((vector_size(16)));
typedef __attribute__((address_space(1))) v4i_t as1_v4i;
typedef __attribute__((address_space(3))) v4i_t as3_v4i;
#define ASYNC_COPY16(gp, lp) \
  __builtin_amdgcn_global_load_async_to_lds_b128((as1_v4i*)(gp), (as3_v4i*)(lp), 0, 0)
#define WAIT_ASYNC() __builtin_amdgcn_s_wait_asynccnt(0)
#else
#define ASYNC_COPY16(gp, lp) (*(uint4*)(lp) = *(const uint4*)(gp))
#define WAIT_ASYNC() ((void)0)
#endif

// ---------------- bf16 split helpers (RNE) ----------------
__device__ __forceinline__ unsigned short bf16_rne(float f) {
  union { float f; uint32_t u; } c; c.f = f;
  uint32_t u = c.u;
  uint32_t r = u + 0x7FFFu + ((u >> 16) & 1u);
  return (unsigned short)(r >> 16);
}
__device__ __forceinline__ float bf16_to_f(unsigned short h) {
  union { uint32_t u; float f; } c; c.u = ((uint32_t)h) << 16;
  return c.f;
}

// ---------------- kernel 1: f32 -> (hi,lo) bf16 planes ----------------
__global__ __launch_bounds__(256) void k_convert(const float4* __restrict__ src,
                                                 ushort4* __restrict__ hi,
                                                 ushort4* __restrict__ lo, int n4) {
  int i = blockIdx.x * 256 + threadIdx.x;
  if (i >= n4) return;
  float4 v = src[i];
  ushort4 h, l;
  h.x = bf16_rne(v.x); l.x = bf16_rne(v.x - bf16_to_f(h.x));
  h.y = bf16_rne(v.y); l.y = bf16_rne(v.y - bf16_to_f(h.y));
  h.z = bf16_rne(v.z); l.z = bf16_rne(v.z - bf16_to_f(h.z));
  h.w = bf16_rne(v.w); l.w = bf16_rne(v.w - bf16_to_f(h.w));
  hi[i] = h; lo[i] = l;
}

// ---------------- kernel 2: hash code table [VV][4] ----------------
__global__ __launch_bounds__(256) void k_hash(const int* __restrict__ a,
                                              const int* __restrict__ b,
                                              const int* __restrict__ p,
                                              int4* __restrict__ codes) {
  int v = blockIdx.x * 256 + threadIdx.x;
  if (v >= VV) return;
  unsigned uv = (unsigned)v;
  int4 c;
  c.x = (int)((((uv * (unsigned)a[0]) + (unsigned)b[0]) % (unsigned)p[0]) & (VCC - 1u));
  c.y = (int)((((uv * (unsigned)a[1]) + (unsigned)b[1]) % (unsigned)p[1]) & (VCC - 1u));
  c.z = (int)((((uv * (unsigned)a[2]) + (unsigned)b[2]) % (unsigned)p[2]) & (VCC - 1u));
  c.w = (int)((((uv * (unsigned)a[3]) + (unsigned)b[3]) % (unsigned)p[3]) & (VCC - 1u));
  codes[v] = c;
}

// ---------------- kernel 3: split-precision bf16 WMMA GEMM ----------------
// logits[m][n] = sum_k x[m][k] * W[n][k]
// WG tile 128x128; 8 waves in 4(M) x 2(N); each wave a 32x64 block (2x4 subtiles).
// Double-buffered LDS filled via async global->LDS copies (ASYNCcnt).

union FragU { uint4 q[2]; v16bf v; };

__device__ __forceinline__ v8f bfwmma(v16bf a, v16bf b, v8f c) {
  return __builtin_amdgcn_wmma_f32_16x16x32_bf16(false, a, false, b, (short)0, c,
                                                 false, false);
}

__device__ __forceinline__ void store_ctile(float* __restrict__ logits, v8f acc,
                                            int row0, int col0, int lane) {
  const int mm = row0 + ((lane >> 4) << 3);   // lanes 16-31 hold M+8
  const int nn = col0 + (lane & 15);
#pragma unroll
  for (int j = 0; j < 8; ++j)
    logits[(size_t)(mm + j) * VCC + nn] = acc[j];
}

__global__ __launch_bounds__(256) void k_gemm(const unsigned short* __restrict__ xh,
                                              const unsigned short* __restrict__ xl,
                                              const unsigned short* __restrict__ wh,
                                              const unsigned short* __restrict__ wl,
                                              float* __restrict__ logits) {
  // 2 buffers x 128 rows x 32 cols bf16, 4 planes = 64 KB total
  __shared__ __align__(16) unsigned short sAh[2 * 128 * LDA];
  __shared__ __align__(16) unsigned short sAl[2 * 128 * LDA];
  __shared__ __align__(16) unsigned short sBh[2 * 128 * LDA];
  __shared__ __align__(16) unsigned short sBl[2 * 128 * LDA];

  const int tid  = threadIdx.x;
  const int lane = tid & 31;
  const int wave = tid >> 5;
  const int wm   = wave & 3;            // 0..3 -> 32-row band
  const int wn   = wave >> 2;           // 0..1 -> 64-col band
  const int m0   = blockIdx.y * 128;
  const int n0   = blockIdx.x * 128;

  // cooperative fill: each thread copies 16 B from rows (lr) and (lr+64) per plane
  const int lr = tid >> 2;              // 0..63
  const int lc = (tid & 3) * 8;         // ushort col: 0,8,16,24

  // fragment indexing (per ISA VGPR layouts)
  const int frow = lane & 15;
  const int fka  = (lane >> 4) * 8;     // A: chunks [fka..+7] and [fka+16..+23]
  const int fkb  = (lane >> 4) * 8;     // B: chunks [fkb..+7] and [fkb+16..+23]

  auto loadTile = [&](int buf, int kk) {
    const int o  = buf * (128 * LDA);
    const int a0 = (m0 + lr) * DD + kk + lc;
    const int b0 = (n0 + lr) * DD + kk + lc;
    const int l0 = o + lr * LDA + lc;
    ASYNC_COPY16(&xh[a0],           &sAh[l0]);
    ASYNC_COPY16(&xh[a0 + 64 * DD], &sAh[l0 + 64 * LDA]);
    ASYNC_COPY16(&xl[a0],           &sAl[l0]);
    ASYNC_COPY16(&xl[a0 + 64 * DD], &sAl[l0 + 64 * LDA]);
    ASYNC_COPY16(&wh[b0],           &sBh[l0]);
    ASYNC_COPY16(&wh[b0 + 64 * DD], &sBh[l0 + 64 * LDA]);
    ASYNC_COPY16(&wl[b0],           &sBl[l0]);
    ASYNC_COPY16(&wl[b0 + 64 * DD], &sBl[l0 + 64 * LDA]);
  };

  const v8f vzero = {0.f, 0.f, 0.f, 0.f, 0.f, 0.f, 0.f, 0.f};
  v8f acc[2][4];
#pragma unroll
  for (int i = 0; i < 2; ++i)
#pragma unroll
    for (int j = 0; j < 4; ++j) acc[i][j] = vzero;

  loadTile(0, 0);                       // prologue prefetch
  for (int kk = 0; kk < DD; kk += KT) {
    const int buf = (kk >> 5) & 1;
    WAIT_ASYNC();                       // own async copies of current tile done
    __syncthreads();                    // whole tile visible to all waves
    if (kk + KT < DD) loadTile(buf ^ 1, kk + KT);   // prefetch next tile

    const int o = buf * (128 * LDA);
    FragU bh[4], bl[4];
#pragma unroll
    for (int sn = 0; sn < 4; ++sn) {
      const int c = o + (wn * 64 + sn * 16 + frow) * LDA + fkb;
      bh[sn].q[0] = *(const uint4*)&sBh[c];
      bh[sn].q[1] = *(const uint4*)&sBh[c + 16];
      bl[sn].q[0] = *(const uint4*)&sBl[c];
      bl[sn].q[1] = *(const uint4*)&sBl[c + 16];
    }
#pragma unroll
    for (int sm = 0; sm < 2; ++sm) {
      const int r = o + (wm * 32 + sm * 16 + frow) * LDA + fka;
      FragU ah, al;
      ah.q[0] = *(const uint4*)&sAh[r];
      ah.q[1] = *(const uint4*)&sAh[r + 16];
      al.q[0] = *(const uint4*)&sAl[r];
      al.q[1] = *(const uint4*)&sAl[r + 16];
      // split-precision product: hi*hi + hi*lo + lo*hi (~16 mantissa bits)
#pragma unroll
      for (int sn = 0; sn < 4; ++sn) {
        acc[sm][sn] = bfwmma(ah.v, bh[sn].v, acc[sm][sn]);
        acc[sm][sn] = bfwmma(ah.v, bl[sn].v, acc[sm][sn]);
        acc[sm][sn] = bfwmma(al.v, bh[sn].v, acc[sm][sn]);
      }
    }
    // no trailing barrier needed: next iteration's wait+barrier orders reuse
  }

#pragma unroll
  for (int sm = 0; sm < 2; ++sm)
#pragma unroll
    for (int sn = 0; sn < 4; ++sn)
      store_ctile(logits, acc[sm][sn],
                  m0 + wm * 32 + sm * 16, n0 + wn * 64 + sn * 16, lane);
}

// ---------------- kernel 4: per-row gather + mean (output-BW bound) ----------------
__global__ __launch_bounds__(256) void k_gather(const float* __restrict__ logits,
                                                const int4* __restrict__ codes,
                                                float* __restrict__ out) {
  __shared__ __align__(16) float srow[VCC];   // 32 KB logits row
  const int r = blockIdx.x;                   // (b,s) flat, 0..2047
  const float4* s4 = (const float4*)(logits + (size_t)r * VCC);
#pragma unroll
  for (int i = threadIdx.x; i < VCC / 4; i += 256)
    ASYNC_COPY16(s4 + i, ((float4*)srow) + i);
  WAIT_ASYNC();
  __syncthreads();

  float* orow = out + (size_t)r * VV;
  for (int v = threadIdx.x; v < VV; v += 256) {
    int4 c = codes[v];
    float m = 0.25f * (srow[c.x] + srow[c.y] + srow[c.z] + srow[c.w]);
    __builtin_nontemporal_store(m, &orow[v]);   // 262 MB stream: don't pollute L2
  }
}

// ---------------- launch ----------------
extern "C" void kernel_launch(void* const* d_in, const int* in_sizes, int n_in,
                              void* d_out, int out_size, void* d_ws, size_t ws_size,
                              hipStream_t stream) {
  (void)in_sizes; (void)n_in; (void)out_size; (void)ws_size;
  const float* x = (const float*)d_in[0];
  const float* W = (const float*)d_in[1];
  const int*   p = (const int*)d_in[2];
  const int*   a = (const int*)d_in[3];
  const int*   b = (const int*)d_in[4];
  // d_in[5] (vocab_size) baked in as VV
  float* out = (float*)d_out;

  // workspace layout (all 16B aligned): ~104.5 MB
  char* ws = (char*)d_ws;
  size_t off = 0;
  float* logits = (float*)(ws + off);            off += (size_t)MM * VCC * sizeof(float);   // 64 MB
  int4*  codes  = (int4*)(ws + off);             off += (size_t)VV * sizeof(int4);          // 512 KB
  unsigned short* xh = (unsigned short*)(ws + off); off += (size_t)MM * DD * 2;             // 4 MB
  unsigned short* xl = (unsigned short*)(ws + off); off += (size_t)MM * DD * 2;             // 4 MB
  unsigned short* wh = (unsigned short*)(ws + off); off += (size_t)VCC * DD * 2;            // 16 MB
  unsigned short* wl = (unsigned short*)(ws + off); off += (size_t)VCC * DD * 2;            // 16 MB

  // 1) split f32 -> bf16 hi/lo planes
  {
    int n4 = MM * DD / 4;     // 524288
    k_convert<<<n4 / 256, 256, 0, stream>>>((const float4*)x, (ushort4*)xh, (ushort4*)xl, n4);
    n4 = VCC * DD / 4;        // 2097152
    k_convert<<<n4 / 256, 256, 0, stream>>>((const float4*)W, (ushort4*)wh, (ushort4*)wl, n4);
  }
  // 2) hash table
  k_hash<<<(VV + 255) / 256, 256, 0, stream>>>(a, b, p, codes);
  // 3) WMMA GEMM: grid = (N/128, M/128)
  k_gemm<<<dim3(VCC / 128, MM / 128), 256, 0, stream>>>(xh, xl, wh, wl, logits);
  // 4) gather + mean: one WG per output row
  k_gather<<<MM, 256, 0, stream>>>(logits, codes, out);
}
